// SpMiddleResNetFHDFocal_32298154066280
// MI455X (gfx1250) — compile-verified, matrix operand order checked by reading, not run on hardware
//
#include <hip/hip_runtime.h>
#include <hip/hip_bf16.h>

// ---------------------------------------------------------------------------
// SpMiddleResNetFHD (dense) forward on MI455X / gfx1250.
//
// Implicit-GEMM conv on V_WMMA_F32_16X16X32_BF16, fp32 accumulate.
// ~100 GFLOP, working set resident in 192MB L2 -> compute bound -> matrix pipes.
//
// Key layout decision: activations live in bf16 channel-minor [S][C] between
// layers, weights are pre-repacked to bf16 [Cout][KDHW][CinP] (tap-major,
// channel-minor). A lane's 8 consecutive K elements are then 8 contiguous
// bf16 -> every fragment is built from b128 loads with ZERO in-loop format
// conversion. Stem is zero-padded 17->32 channels so all CIN are multiples
// of 32 (no K-tail anywhere).
//
// Per wave: 16(cout) x 64(spatial) tile = 4 WMMAs per K-step sharing one A
// fragment. Inner loop = 10x global_load_b128 + 4x v_wmma (+ cndmask on
// boundary taps only).
// ---------------------------------------------------------------------------

typedef __attribute__((ext_vector_type(16))) __bf16 v16bf;
typedef __attribute__((ext_vector_type(8)))  __bf16 v8bf;
typedef __attribute__((ext_vector_type(8)))  float  v8f;

static __device__ __forceinline__ v16bf joinbf(v8bf a, v8bf b) {
    return __builtin_shufflevector(a, b, 0,1,2,3,4,5,6,7,8,9,10,11,12,13,14,15);
}

// ---------------------------------------------------------------------------
// Repack kernels (run once per launch, before the conv chain)
// ---------------------------------------------------------------------------

// x: fp32 [17][S] -> bf16 [S][32], channels 17..31 zero.
__global__ __launch_bounds__(256) void pack_x(const float* __restrict__ X,
                                              __bf16* __restrict__ O) {
    constexpr int S = 8 * 56 * 56;
    const int i = blockIdx.x * 256 + threadIdx.x;   // over S*32
    if (i >= S * 32) return;
    const int c = i & 31;
    const int n = i >> 5;
    const float v = (c < 17) ? X[(size_t)c * S + n] : 0.0f;
    O[i] = (__bf16)v;
}

// W: fp32 [Cout][Cin][KDHW] -> bf16 [Cout][KDHW][CinP], channels >= Cin zero.
__global__ __launch_bounds__(256) void pack_w(const float* __restrict__ W,
                                              __bf16* __restrict__ O,
                                              int Cout, int Cin, int CinP, int KDHW) {
    const long long N = (long long)Cout * KDHW * CinP;
    const long long i = (long long)blockIdx.x * 256 + threadIdx.x;
    if (i >= N) return;
    const int ci = (int)(i % CinP);
    const int t  = (int)((i / CinP) % KDHW);
    const int co = (int)(i / ((long long)CinP * KDHW));
    const float v = (ci < Cin) ? W[((size_t)co * Cin + ci) * KDHW + t] : 0.0f;
    O[i] = (__bf16)v;
}

// ---------------------------------------------------------------------------
// Fused conv + BN (+res)(+relu) implicit GEMM.
// Fragment layouts per CDNA5 ISA 7.12.2 (wave32):
//   A (16x32 bf16): lane l holds row M=l%16; element e -> K=(e<8?e:e+8)+8*(l>=16)
//   B (32x16 bf16): lane l holds col N=l%16; same K mapping (mirrored)
//   C/D (16x16 f32): vgpr r -> M = r + 8*(l>=16), N = l%16
// ---------------------------------------------------------------------------
template<int KD, int KH, int KW, int PD, int PH, int PW, int CINP,
         bool HASRES, bool RELU>
__global__ __launch_bounds__(256)
void conv_bn_wmma(const __bf16* __restrict__ Xbf,  // [S][CINP]
                  const __bf16* __restrict__ Wbf,  // [Cout][KDHW][CINP]
                  const float* __restrict__ gg, const float* __restrict__ bb,
                  const float* __restrict__ mm, const float* __restrict__ vv,
                  float eps,
                  const __bf16* resbf,             // [S][Cout] (may alias Ybf)
                  __bf16* Ybf,                     // [S][Cout] or null
                  float* __restrict__ Yf32,        // [Cout][S] or null (final layer)
                  int Cout)
{
    constexpr int D = 8, H = 56, W = 56;
    constexpr int S = D * H * W;              // 25088
    constexpr int KHW  = KH * KW;
    constexpr int KDHW = KD * KH * KW;
    constexpr int NT = 4;                     // 16-wide N tiles per wave
    constexpr int nQuad = S / (16 * NT);      // 392
    constexpr bool EDGE = (KD > 1 || KH > 1 || KW > 1 || PD > 0 || PH > 0 || PW > 0);

    const int wave  = blockIdx.x * (blockDim.x >> 5) + (threadIdx.x >> 5);
    const int tiles = (Cout >> 4) * nQuad;
    if (wave >= tiles) return;                // wave-uniform -> EXEC all-ones for WMMA

    const int lane = threadIdx.x & 31;
    const int lmod = lane & 15;
    const int half = lane >> 4;
    const int koffBase = half << 3;

    const int coT = wave / nQuad;
    const int nT  = wave - coT * nQuad;
    const int coBase = coT << 4;
    const int nBase  = nT * (16 * NT);

    int spN[NT], dd[NT], hhv[NT], wwv[NT];
#pragma unroll
    for (int j = 0; j < NT; ++j) {
        spN[j] = nBase + lmod + 16 * j;
        int t = spN[j];
        wwv[j] = t % W; t /= W;
        hhv[j] = t % H;
        dd[j]  = t / H;
    }

    // A row pointer: weight layout [Cout][KDHW][CINP], K enumerated (tap, ci).
    const __bf16* __restrict__ wRow =
        Wbf + (size_t)(coBase + lmod) * (KDHW * CINP) + koffBase;

    v8f acc[NT] = {};
    const v8bf zbf = {};

    for (int tap = 0; tap < KDHW; ++tap) {    // compile-time trip count (1/3/9/49)
        const int kd = tap / KHW;
        const int kh = (tap % KHW) / KW;
        const int kw = tap % KW;

        // Per-tap, per-column validity + clamped base pointer (branch-free).
        const __bf16* xp[NT];
        bool ok[NT];
#pragma unroll
        for (int j = 0; j < NT; ++j) {
            int id = dd[j]  + kd - PD;
            int ih = hhv[j] + kh - PH;
            int iw = wwv[j] + kw - PW;
            bool o = true;
            if (KD > 1 || PD > 0) { o = o && ((unsigned)id < (unsigned)D);
                                    id = id < 0 ? 0 : (id > D - 1 ? D - 1 : id); }
            if (KH > 1 || PH > 0) { o = o && ((unsigned)ih < (unsigned)H);
                                    ih = ih < 0 ? 0 : (ih > H - 1 ? H - 1 : ih); }
            if (KW > 1 || PW > 0) { o = o && ((unsigned)iw < (unsigned)W);
                                    iw = iw < 0 ? 0 : (iw > W - 1 ? W - 1 : iw); }
            ok[j] = o;
            xp[j] = Xbf + (size_t)((id * H + ih) * W + iw) * CINP + koffBase;
        }

        const __bf16* wTap = wRow + tap * CINP;

        for (int c0 = 0; c0 < CINP; c0 += 32) {
            // A fragment: two b128 loads (contiguous bf16 along K).
            const v8bf a0 = *(const v8bf*)(wTap + c0);
            const v8bf a1 = *(const v8bf*)(wTap + c0 + 16);
            const v16bf af = joinbf(a0, a1);

            v16bf bf[NT];
#pragma unroll
            for (int j = 0; j < NT; ++j) {
                v8bf b0 = *(const v8bf*)(xp[j] + c0);        // always legal (clamped)
                v8bf b1 = *(const v8bf*)(xp[j] + c0 + 16);
                if (EDGE) {                                  // folds away for 1x1x1
                    b0 = ok[j] ? b0 : zbf;
                    b1 = ok[j] ? b1 : zbf;
                }
                bf[j] = joinbf(b0, b1);
            }

#pragma unroll
            for (int j = 0; j < NT; ++j)
                acc[j] = __builtin_amdgcn_wmma_f32_16x16x32_bf16(
                    false, af, false, bf[j], (short)0, acc[j], false, false);
        }
    }

    // ---- Epilogue: BN fold, optional residual, optional ReLU ----------------
    const int coStore = coBase + (half << 3);     // lane's 8 contiguous channels
    float inv[8], beta[8];
#pragma unroll
    for (int r = 0; r < 8; ++r) {
        const int co = coStore + r;
        inv[r]  = gg[co] * rsqrtf(vv[co] + eps);
        beta[r] = bb[co] - mm[co] * inv[r];
    }

#pragma unroll
    for (int j = 0; j < NT; ++j) {
        float y[8];
        v8bf rb = {};
        if (HASRES)
            rb = *(const v8bf*)(resbf + (size_t)spN[j] * Cout + coStore);
#pragma unroll
        for (int r = 0; r < 8; ++r) {
            float t = acc[j][r] * inv[r] + beta[r];
            if (HASRES) t += (float)rb[r];
            if (RELU)   t = fmaxf(t, 0.0f);
            y[r] = t;
        }
        if (Ybf) {                                // uniform branch
            v8bf ov;
#pragma unroll
            for (int r = 0; r < 8; ++r) ov[r] = (__bf16)y[r];
            *(v8bf*)(Ybf + (size_t)spN[j] * Cout + coStore) = ov;
        }
        if (Yf32) {                               // uniform branch (final layer)
#pragma unroll
            for (int r = 0; r < 8; ++r)
                Yf32[(size_t)(coStore + r) * S + spN[j]] = y[r];
        }
    }
}

// ---------------------------------------------------------------------------
// Host side
// ---------------------------------------------------------------------------

struct BNP { const float *g, *b, *m, *v; };
struct Blk {
    const float *w1, *w2, *w3, *wd;
    __bf16 *w1b, *w2b, *w3b, *wdb;
    BNP bn1, bn2, bn3, bnd;
    int cin, mid, cout;
    bool down, k311;
};

template<int KD, int KH, int KW, int PD, int PH, int PW, int CINP,
         bool HASRES, bool RELU>
static inline void run_conv(hipStream_t st, const __bf16* X, const __bf16* W,
                            BNP bn, float eps, const __bf16* res,
                            __bf16* Ybf, float* Yf32, int Cout)
{
    constexpr int S = 8 * 56 * 56;
    const int tiles   = (Cout / 16) * (S / 64);
    const int nblocks = (tiles + 7) / 8;          // 8 waves (256 thr) per block
    conv_bn_wmma<KD, KH, KW, PD, PH, PW, CINP, HASRES, RELU>
        <<<nblocks, 256, 0, st>>>(X, W, bn.g, bn.b, bn.m, bn.v, eps,
                                  res, Ybf, Yf32, Cout);
}

// conv1 of c1 blocks: 1x1x1, relu, no res
static inline void conv111_relu(hipStream_t st, const __bf16* X, const __bf16* W,
                                BNP bn, __bf16* Y, int Cin, int Cout) {
    if (Cin == 32) run_conv<1,1,1,0,0,0, 32, false, true>(st, X, W, bn, 1e-3f, nullptr, Y, nullptr, Cout);
    else           run_conv<1,1,1,0,0,0, 128,false, true>(st, X, W, bn, 1e-3f, nullptr, Y, nullptr, Cout);
}
// downsample: 1x1x1, no relu, no res
static inline void conv111_plain(hipStream_t st, const __bf16* X, const __bf16* W,
                                 BNP bn, __bf16* Y, int Cin, int Cout) {
    switch (Cin) {
        case 32:  run_conv<1,1,1,0,0,0, 32, false, false>(st, X, W, bn, 1e-3f, nullptr, Y, nullptr, Cout); break;
        case 128: run_conv<1,1,1,0,0,0, 128,false, false>(st, X, W, bn, 1e-3f, nullptr, Y, nullptr, Cout); break;
        default:  run_conv<1,1,1,0,0,0, 256,false, false>(st, X, W, bn, 1e-3f, nullptr, Y, nullptr, Cout); break;
    }
}
// conv3: 1x1x1, relu, residual; optional fp32 final output
static inline void conv111_res(hipStream_t st, const __bf16* X, const __bf16* W,
                               BNP bn, const __bf16* res, __bf16* Ybf, float* Yf32,
                               int Cin, int Cout) {
    switch (Cin) {
        case 32:  run_conv<1,1,1,0,0,0, 32, true, true>(st, X, W, bn, 1e-3f, res, Ybf, Yf32, Cout); break;
        case 64:  run_conv<1,1,1,0,0,0, 64, true, true>(st, X, W, bn, 1e-3f, res, Ybf, Yf32, Cout); break;
        default:  run_conv<1,1,1,0,0,0, 128,true, true>(st, X, W, bn, 1e-3f, res, Ybf, Yf32, Cout); break;
    }
}
// conv1 of c2/c3: 3x1x1 pad(1,0,0), relu, no res
static inline void conv311_relu(hipStream_t st, const __bf16* X, const __bf16* W,
                                BNP bn, __bf16* Y, int Cin, int Cout) {
    switch (Cin) {
        case 128: run_conv<3,1,1,1,0,0, 128, false, true>(st, X, W, bn, 1e-3f, nullptr, Y, nullptr, Cout); break;
        case 256: run_conv<3,1,1,1,0,0, 256, false, true>(st, X, W, bn, 1e-3f, nullptr, Y, nullptr, Cout); break;
        default:  run_conv<3,1,1,1,0,0, 512, false, true>(st, X, W, bn, 1e-3f, nullptr, Y, nullptr, Cout); break;
    }
}
// conv2: 1x3x3 pad(0,1,1), no relu, no res
static inline void conv133_plain(hipStream_t st, const __bf16* X, const __bf16* W,
                                 BNP bn, __bf16* Y, int Cin, int Cout) {
    switch (Cin) {
        case 32:  run_conv<1,3,3,0,1,1, 32, false, false>(st, X, W, bn, 1e-3f, nullptr, Y, nullptr, Cout); break;
        case 64:  run_conv<1,3,3,0,1,1, 64, false, false>(st, X, W, bn, 1e-3f, nullptr, Y, nullptr, Cout); break;
        default:  run_conv<1,3,3,0,1,1, 128,false, false>(st, X, W, bn, 1e-3f, nullptr, Y, nullptr, Cout); break;
    }
}

extern "C" void kernel_launch(void* const* d_in, const int* in_sizes, int n_in,
                              void* d_out, int out_size, void* d_ws, size_t ws_size,
                              hipStream_t stream)
{
    (void)n_in; (void)out_size; (void)ws_size;

    const long long S = 8LL * 56 * 56;

    // --- block topology ------------------------------------------------------
    Blk blk[13]; int nb = 0;
    auto addStage = [&](int n, int cin0, int mid, int cout, bool k311) {
        for (int i = 0; i < n; ++i) {
            Blk B{}; B.cin = (i == 0) ? cin0 : cout; B.mid = mid; B.cout = cout;
            B.down = (i == 0); B.k311 = k311; blk[nb++] = B;
        }
    };
    addStage(4,  32,  32, 128, false);  // c1: conv1 = 1x1x1
    addStage(6, 128,  64, 256, true);   // c2: conv1 = 3x1x1, pad (1,0,0)
    addStage(3, 256, 128, 512, true);   // c3: conv1 = 3x1x1, pad (1,0,0)

    // --- decode flattened input list (two plausible flattening orders) -------
    int cur = 0;
    auto nx  = [&]() { return (const float*)d_in[cur++]; };
    auto nbn = [&]() { BNP p; p.g = nx(); p.b = nx(); p.m = nx(); p.v = nx(); return p; };

    const float* x = nullptr;
    const float* inpW = nullptr; BNP inpBN{};
    const bool insertOrder = (in_sizes[0] == 17 * 8 * 56 * 56);

    if (insertOrder) {           // dict insertion order: x, inp{w,bn}, blocks{w1,bn1,...}
        x = nx();
        inpW = nx(); inpBN = nbn();
        for (int i = 0; i < nb; ++i) {
            Blk& B = blk[i];
            B.w1 = nx(); B.bn1 = nbn();
            B.w2 = nx(); B.bn2 = nbn();
            B.w3 = nx(); B.bn3 = nbn();
            if (B.down) { B.wd = nx(); B.bnd = nbn(); }
        }
    } else {                     // JAX pytree sorted keys: c1,c2,c3,inp then x
        for (int i = 0; i < nb; ++i) {
            Blk& B = blk[i];
            B.bn1 = nbn(); B.bn2 = nbn(); B.bn3 = nbn();
            if (B.down) B.bnd = nbn();
            B.w1 = nx(); B.w2 = nx(); B.w3 = nx();
            if (B.down) B.wd = nx();
        }
        inpBN = nbn(); inpW = nx();
        x = nx();
    }

    // --- workspace layout (bf16 elements), ~72 MB total ----------------------
    __bf16* wsb = (__bf16*)d_ws;
    size_t wcur = 0;
    auto walloc = [&](size_t nelem) {
        __bf16* p = wsb + wcur;
        wcur += (nelem + 7) & ~(size_t)7;      // keep 16B alignment
        return p;
    };
    __bf16* xpad = walloc((size_t)32  * S);    // stem input, padded to 32ch
    __bf16* bP   = walloc((size_t)512 * S);    // ping
    __bf16* bQ   = walloc((size_t)512 * S);    // pong
    __bf16* bM1  = walloc((size_t)128 * S);    // mid1
    __bf16* bM2  = walloc((size_t)128 * S);    // mid2

    auto packw = [&](const float* W, int Cout, int Cin, int CinP, int kdhw) {
        const size_t n = (size_t)Cout * kdhw * CinP;
        __bf16* p = walloc(n);
        pack_w<<<(int)((n + 255) / 256), 256, 0, stream>>>(W, p, Cout, Cin, CinP, kdhw);
        return p;
    };

    // Repack all weights (bf16, tap-major/channel-minor, channel-padded).
    __bf16* inpWb = packw(inpW, 32, 17, 32, 49);
    for (int i = 0; i < nb; ++i) {
        Blk& B = blk[i];
        B.w1b = packw(B.w1, B.mid,  B.cin, B.cin, B.k311 ? 3 : 1);
        B.w2b = packw(B.w2, B.mid,  B.mid, B.mid, 9);
        B.w3b = packw(B.w3, B.cout, B.mid, B.mid, 1);
        if (B.down) B.wdb = packw(B.wd, B.cout, B.cin, B.cin, 1);
    }

    // Transpose/convert input: fp32 [17][S] -> bf16 [S][32] (zero-padded).
    pack_x<<<(int)((32 * S + 255) / 256), 256, 0, stream>>>(x, xpad);

    // --- stem: 17(->32) -> 32, k(1,7,7), pad(0,3,3), bn(eps=1e-5), relu ------
    run_conv<1, 7, 7, 0, 3, 3, 32, false, true>(stream, xpad, inpWb, inpBN, 1e-5f,
                                                nullptr, bP, nullptr, 32);

    // --- residual blocks -----------------------------------------------------
    for (int i = 0; i < nb; ++i) {
        const Blk& B = blk[i];
        const bool last = (i == nb - 1);

        // conv1 -> bn1 -> relu
        if (B.k311) conv311_relu(stream, bP, B.w1b, B.bn1, bM1, B.cin, B.mid);
        else        conv111_relu(stream, bP, B.w1b, B.bn1, bM1, B.cin, B.mid);

        // conv2 (1,3,3) pad (0,1,1) -> bn2
        conv133_plain(stream, bM1, B.w2b, B.bn2, bM2, B.mid, B.mid);

        // identity path
        const __bf16* resPtr;
        if (B.down) {
            // idn = bnd(convd(x)) into bQ; conv3 then does a same-index
            // read-add-write in place (one lane per 16B region -> race free).
            conv111_plain(stream, bP, B.wdb, B.bnd, bQ, B.cin, B.cout);
            resPtr = bQ;
        } else {
            resPtr = bP;
        }

        // conv3 -> bn3, + identity, relu.
        // Last layer: also emit fp32 NCDHW straight into d_out.
        __bf16* ybf = last ? nullptr : bQ;
        float*  yf  = last ? (float*)d_out : nullptr;
        conv111_res(stream, bM2, B.w3b, B.bn3, resPtr, ybf, yf, B.mid, B.cout);

        if (!last) { __bf16* t = bP; bP = bQ; bQ = t; }
    }
}
